// InterpretableMultiHeadAttention_18030272708888
// MI455X (gfx1250) — matrix operand order repaired and checked by reading, not compiled, not run on hardware
//
#include <hip/hip_runtime.h>
#include <cmath>

typedef __attribute__((ext_vector_type(16))) _Float16 v16h;
typedef __attribute__((ext_vector_type(8)))  _Float16 v8h;
typedef __attribute__((ext_vector_type(8)))  float    v8f;

#define WMMA_F16(a, b, c) \
  __builtin_amdgcn_wmma_f32_16x16x32_f16(false, (a), false, (b), (short)0, (c), false, false)

static constexpr int Bn = 4;     // batch
static constexpr int Sn = 1024;  // sequence length
static constexpr int Dn = 1024;  // d_model
static constexpr int Hn = 8;     // heads
static constexpr int Kn = 128;   // d_k = d_v

// ---------------- fp32 -> fp16 convert ----------------
__global__ void cvt_f32_f16_kernel(const float* __restrict__ src,
                                   _Float16* __restrict__ dst, int n) {
  int i = blockIdx.x * blockDim.x + threadIdx.x;
  int stride = gridDim.x * blockDim.x;
  for (; i < n; i += stride) dst[i] = (_Float16)src[i];
}

// fp32 [z][R][C] -> fp16 [z][C][R]  (weight transpose; coalesced reads)
__global__ void cvt_f32_f16_t_kernel(const float* __restrict__ src,
                                     _Float16* __restrict__ dst, int R, int C) {
  const float* s = src + (size_t)blockIdx.z * R * C;
  _Float16* d = dst + (size_t)blockIdx.z * R * C;
  int i = blockIdx.x * blockDim.x + threadIdx.x;
  int stride = gridDim.x * blockDim.x;
  for (; i < R * C; i += stride) {
    int r = i / C, c = i - r * C;
    d[(size_t)c * R + r] = (_Float16)s[i];
  }
}

// concat two 8-wide f16 vectors into the 16-wide WMMA fragment
__device__ __forceinline__ v16h cat8(v8h lo, v8h hi) {
  return __builtin_shufflevector(lo, hi, 0, 1, 2, 3, 4, 5, 6, 7, 8, 9, 10, 11,
                                 12, 13, 14, 15);
}

// A-fragment from a row-major f16 row (ISA 7.12.2 16-bit A layout):
// lane half 0: K = 0..7 and 16..23; half 1: K = 8..15 and 24..31.
__device__ __forceinline__ v16h load_a(const _Float16* __restrict__ row_k0,
                                       int half) {
  v8h s0 = *(const v8h*)(row_k0 + half * 8);
  v8h s1 = *(const v8h*)(row_k0 + 16 + half * 8);
  return cat8(s0, s1);
}

// B-fragment from transposed-B storage (BT row n = B column n, K contiguous):
// lane half 0: K = 0..7 and 16..23; half 1: K = 8..15 and 24..31 (matches A).
__device__ __forceinline__ v16h load_bt(const _Float16* __restrict__ bt_row_k0,
                                        int half) {
  v8h s0 = *(const v8h*)(bt_row_k0 + half * 8);
  v8h s1 = *(const v8h*)(bt_row_k0 + 16 + half * 8);
  return cat8(s0, s1);
}

// ------------- unified GEMM:  C[M,N] = alpha * A[M,K] * BT[N,K]^T -----------
// A row-major (lda); BT row-major (ldbt) = B stored column-major; batched over
// blockIdx.z. Block = 128 threads = 4 waves. Wave tile = (16*MW) x 64
// (MW a-fragments, 4*MW v8f accumulators). No LDS, no barriers: all fragments
// are contiguous global_load_b128 pairs.
// CT=true writes C transposed ([N,M], ldc = rows of M-space): contiguous 16B
// stores per lane (used to produce vs^T for the attention*V stage).
template <typename TOUT, int MW, bool CT>
__global__ __launch_bounds__(128) void gemm_tn_kernel(
    const _Float16* __restrict__ A, int lda, long long Abstride,
    const _Float16* __restrict__ BT, int ldbt, long long Bbstride,
    TOUT* __restrict__ C, int ldc, long long Cbstride, int K, float alpha) {
  const int tid = threadIdx.x;
  const int wave = tid >> 5;
  const int lane = tid & 31;
  const int half = lane >> 4;
  const int ln = lane & 15;

  const _Float16* Ab = A + (size_t)blockIdx.z * (size_t)Abstride;
  const _Float16* Bb = BT + (size_t)blockIdx.z * (size_t)Bbstride;
  TOUT* Cb = C + (size_t)blockIdx.z * (size_t)Cbstride;

  const int n0 = blockIdx.x * 64;
  const int m0 = blockIdx.y * (64 * MW) + wave * (16 * MW);

  v8f acc[MW][4] = {};
  for (int k0 = 0; k0 < K; k0 += 32) {
    v16h bfrag[4];
#pragma unroll
    for (int c = 0; c < 4; ++c)
      bfrag[c] = load_bt(Bb + (size_t)(n0 + c * 16 + ln) * ldbt + k0, half);
#pragma unroll
    for (int w = 0; w < MW; ++w) {
      v16h afrag = load_a(Ab + (size_t)(m0 + w * 16 + ln) * lda + k0, half);
#pragma unroll
      for (int c = 0; c < 4; ++c)
        acc[w][c] = WMMA_F16(afrag, bfrag[c], acc[w][c]);
    }
  }

  // Epilogue. C layout: VGPR r -> m = r + 8*half, lane -> n = ln (per 16-col).
#pragma unroll
  for (int w = 0; w < MW; ++w) {
#pragma unroll
    for (int c = 0; c < 4; ++c) {
      if constexpr (CT) {
        alignas(16) TOUT o[8];
#pragma unroll
        for (int r = 0; r < 8; ++r) o[r] = (TOUT)(acc[w][c][r] * alpha);
        TOUT* dst =
            Cb + (size_t)(n0 + c * 16 + ln) * ldc + m0 + w * 16 + 8 * half;
        *(uint4*)dst = *(const uint4*)o;  // 8 contiguous f16 = 16B store
      } else {
#pragma unroll
        for (int r = 0; r < 8; ++r) {
          const int mm = m0 + w * 16 + r + 8 * half;
          const int nn = n0 + c * 16 + ln;
          Cb[(size_t)mm * ldc + nn] = (TOUT)(acc[w][c][r] * alpha);
        }
      }
    }
  }
}

// ------------- row softmax in place + f16 side copy (row = Sn = 1024) -------
// Writes normalized probabilities as fp32 (the attn output in d_out) AND as
// fp16 into workspace so the attention*V GEMM reads half the bytes and needs
// no in-register conversion.
__global__ __launch_bounds__(256) void softmax_kernel(
    float* __restrict__ attn, _Float16* __restrict__ attn_h) {
  float* p = attn + (size_t)blockIdx.x * Sn;
  _Float16* ph = attn_h + (size_t)blockIdx.x * Sn;
  const int t = threadIdx.x;  // 256 threads, 4 elems each
  float x0 = p[t], x1 = p[t + 256], x2 = p[t + 512], x3 = p[t + 768];

  __shared__ float red[256];
  red[t] = fmaxf(fmaxf(x0, x1), fmaxf(x2, x3));
  __syncthreads();
  for (int s = 128; s > 0; s >>= 1) {
    if (t < s) red[t] = fmaxf(red[t], red[t + s]);
    __syncthreads();
  }
  const float mx = red[0];
  __syncthreads();

  x0 = __expf(x0 - mx); x1 = __expf(x1 - mx);
  x2 = __expf(x2 - mx); x3 = __expf(x3 - mx);
  red[t] = x0 + x1 + x2 + x3;
  __syncthreads();
  for (int s = 128; s > 0; s >>= 1) {
    if (t < s) red[t] += red[t + s];
    __syncthreads();
  }
  const float inv = 1.0f / red[0];
  x0 *= inv; x1 *= inv; x2 *= inv; x3 *= inv;
  p[t] = x0; p[t + 256] = x1; p[t + 512] = x2; p[t + 768] = x3;
  ph[t] = (_Float16)x0; ph[t + 256] = (_Float16)x1;
  ph[t + 512] = (_Float16)x2; ph[t + 768] = (_Float16)x3;
}

// ---------------- pooled = (1/H) * sum_h attn[h,b] @ vs[b] ------------------
// attn read as f16 (softmax side copy); vsT is [DK][B*S] so B-fragments are
// contiguous b128 loads, reused across the 8 heads of each K-step. No LDS.
__global__ __launch_bounds__(128) void heads_kernel(
    const _Float16* __restrict__ attn_h, const _Float16* __restrict__ vsT,
    _Float16* __restrict__ pooled) {
  const int b = blockIdx.z;
  const int tid = threadIdx.x;
  const int wave = tid >> 5;
  const int lane = tid & 31;
  const int half = lane >> 4;
  const int ln = lane & 15;
  const int n0 = blockIdx.x * 64;
  const int m0 = blockIdx.y * 64 + wave * 16;
  const int m = m0 + ln;

  v8f acc[4] = {};
  for (int t0 = 0; t0 < Sn; t0 += 32) {
    v16h bfrag[4];
#pragma unroll
    for (int c = 0; c < 4; ++c)
      bfrag[c] = load_bt(
          vsT + (size_t)(n0 + c * 16 + ln) * (Bn * Sn) + b * Sn + t0, half);

#pragma unroll
    for (int h = 0; h < Hn; ++h) {
      v16h afrag = load_a(
          attn_h + ((size_t)(h * Bn + b) * Sn + m) * Sn + t0, half);
#pragma unroll
      for (int c = 0; c < 4; ++c) acc[c] = WMMA_F16(afrag, bfrag[c], acc[c]);
    }
  }

#pragma unroll
  for (int c = 0; c < 4; ++c) {
#pragma unroll
    for (int r = 0; r < 8; ++r) {
      const int mm = m0 + r + 8 * half;
      const int nn = n0 + c * 16 + ln;
      pooled[(size_t)(b * Sn + mm) * Kn + nn] = (_Float16)(acc[c][r] * 0.125f);
    }
  }
}

// ---------------- host launch ----------------
extern "C" void kernel_launch(void* const* d_in, const int* in_sizes, int n_in,
                              void* d_out, int out_size, void* d_ws,
                              size_t ws_size, hipStream_t stream) {
  (void)in_sizes; (void)n_in; (void)out_size; (void)ws_size;

  const float* q  = (const float*)d_in[0];
  const float* k  = (const float*)d_in[1];
  const float* v  = (const float*)d_in[2];
  const float* Wq = (const float*)d_in[3];
  const float* Wk = (const float*)d_in[4];
  const float* Wv = (const float*)d_in[5];
  const float* Wo = (const float*)d_in[6];

  float* out  = (float*)d_out;                // [B,S,D]
  float* attn = out + (size_t)Bn * Sn * Dn;   // [H,B,S,S]

  // workspace layout (f16 staging, 256B-aligned slots; ~116 MB total)
  char* ws = (char*)d_ws;
  size_t off = 0;
  auto take = [&](size_t elems) {
    _Float16* p = (_Float16*)(ws + off);
    off += (elems * sizeof(_Float16) + 255) & ~(size_t)255;
    return p;
  };
  _Float16* q_h      = take((size_t)Bn * Sn * Dn);
  _Float16* k_h      = take((size_t)Bn * Sn * Dn);
  _Float16* v_h      = take((size_t)Bn * Sn * Dn);
  _Float16* WqT_h    = take((size_t)Hn * Dn * Kn);   // [h][DK][D]
  _Float16* WkT_h    = take((size_t)Hn * Dn * Kn);   // [h][DK][D]
  _Float16* WvT_h    = take((size_t)Dn * Kn);        // [DK][D]
  _Float16* WoT_h    = take((size_t)Kn * Dn);        // [D][DK]
  _Float16* qs_h     = take((size_t)Hn * Bn * Sn * Kn);  // [h][B*S][DK]
  _Float16* ks_h     = take((size_t)Hn * Bn * Sn * Kn);  // [h][B*S][DK]
  _Float16* vsT_h    = take((size_t)Bn * Sn * Kn);       // [DK][B*S]
  _Float16* pooled_h = take((size_t)Bn * Sn * Kn);       // [B*S][DK]
  _Float16* attn_h   = take((size_t)Hn * Bn * Sn * Sn);  // [h][B][S][S] f16

  // fp32 -> fp16 (activations) and fp32 -> fp16 transposed (weights)
  cvt_f32_f16_kernel<<<1024, 256, 0, stream>>>(q, q_h, Bn * Sn * Dn);
  cvt_f32_f16_kernel<<<1024, 256, 0, stream>>>(k, k_h, Bn * Sn * Dn);
  cvt_f32_f16_kernel<<<1024, 256, 0, stream>>>(v, v_h, Bn * Sn * Dn);
  cvt_f32_f16_t_kernel<<<dim3(128, 1, Hn), 256, 0, stream>>>(Wq, WqT_h, Dn, Kn);
  cvt_f32_f16_t_kernel<<<dim3(128, 1, Hn), 256, 0, stream>>>(Wk, WkT_h, Dn, Kn);
  cvt_f32_f16_t_kernel<<<dim3(128, 1, 1), 256, 0, stream>>>(Wv, WvT_h, Dn, Kn);
  cvt_f32_f16_t_kernel<<<dim3(128, 1, 1), 256, 0, stream>>>(Wo, WoT_h, Kn, Dn);

  const dim3 blk(128);

  // qs[h] = q @ Wq[h]   (A shared across heads; BT batched per head)
  gemm_tn_kernel<_Float16, 2, false>
      <<<dim3(Kn / 64, (Bn * Sn) / 128, Hn), blk, 0, stream>>>(
          q_h, Dn, 0LL, WqT_h, Dn, (long long)Dn * Kn, qs_h, Kn,
          (long long)Bn * Sn * Kn, Dn, 1.0f);
  // ks[h] = k @ Wk[h]
  gemm_tn_kernel<_Float16, 2, false>
      <<<dim3(Kn / 64, (Bn * Sn) / 128, Hn), blk, 0, stream>>>(
          k_h, Dn, 0LL, WkT_h, Dn, (long long)Dn * Kn, ks_h, Kn,
          (long long)Bn * Sn * Kn, Dn, 1.0f);
  // vs^T = (v @ Wv)^T   (transposed store -> [DK][B*S])
  gemm_tn_kernel<_Float16, 2, true>
      <<<dim3(Kn / 64, (Bn * Sn) / 128, 1), blk, 0, stream>>>(
          v_h, Dn, 0LL, WvT_h, Dn, 0LL, vsT_h, Bn * Sn, 0LL, Dn, 1.0f);

  // scores[h,b] = (qs @ ks^T) / sqrt(dk): ks rows are already B^T form.
  gemm_tn_kernel<float, 2, false>
      <<<dim3(Sn / 64, Sn / 128, Hn * Bn), blk, 0, stream>>>(
          qs_h, Kn, (long long)Sn * Kn, ks_h, Kn, (long long)Sn * Kn, attn, Sn,
          (long long)Sn * Sn, Kn, 1.0f / sqrtf((float)Kn));

  // softmax in place (fp32 attn output) + fp16 side copy for the next GEMM
  softmax_kernel<<<dim3(Hn * Bn * Sn), dim3(256), 0, stream>>>(attn, attn_h);

  // pooled = mean_h(attn[h] @ vs)
  heads_kernel<<<dim3(Kn / 64, Sn / 64, Bn), blk, 0, stream>>>(attn_h, vsT_h,
                                                               pooled_h);
  // outputs = pooled @ Wo (fp32 out)
  gemm_tn_kernel<float, 2, false>
      <<<dim3(Dn / 64, (Bn * Sn) / 128, 1), blk, 0, stream>>>(
          pooled_h, Kn, 0LL, WoT_h, Kn, 0LL, out, Dn, 0LL, Kn, 1.0f);
}
